// EncoderLayer_15023795601424
// MI455X (gfx1250) — matrix operand order repaired and checked by reading, not compiled, not run on hardware
//
#include <hip/hip_runtime.h>

// ---------------------------------------------------------------------------
// Model constants
// ---------------------------------------------------------------------------
#define DMODEL 1024
#define NHEAD  16
#define DK     64
#define DINNER 4096
#define NBLK   9
#define NSLOT  4
#define BZ     8
#define TLEN   4100
#define NM     8          // NBLK-1
#define BS     512        // TLEN / NM
#define NT     32800      // BZ * TLEN
#define NTF    33587200L  // NT * DMODEL

typedef __attribute__((ext_vector_type(16))) __bf16        bf16x16;
typedef __attribute__((ext_vector_type(8)))  float         floatx8;
typedef __attribute__((ext_vector_type(4)))  float         float4v;
typedef __attribute__((ext_vector_type(4)))  unsigned int  uint4v;

static __device__ __forceinline__ unsigned int pk2(float a, float b) {
    union { __bf16 h[2]; unsigned int u; } t;
    t.h[0] = (__bf16)a; t.h[1] = (__bf16)b;   // -> v_cvt_pk_bf16_f32
    return t.u;
}

// ---------------------------------------------------------------------------
// WMMA GEMM: C[M,N] = act(A[M,K] * W[K,N] + bias), row-major fp32 in/out,
// bf16 multiply / f32 accumulate via v_wmma_f32_16x16x32_bf16.
// Block tile 128x128, K-step 32. 8 waves in 4(M) x 2(N); each wave owns a
// 32x64 tile = 2x4 WMMA fragments = 8 WMMAs per K-step.
// LDS tiles are stored as packed bf16 PAIRS (uint32) in fragment-native
// order so every fragment is exactly two ds_load_b128:
//   As[row][pair]  : pair j holds (k=2j, 2j+1);  lane needs j = half*4+{0..3}
//                    and 8+half*4+{0..3}   (ISA 7.12.2 16-bit A layout)
//   Bs[col][pair]  : pair t holds (k=2t, 2t+1);  lane needs t = half*4+{0..3}, 8+half*4+{0..3}
// Stride 20 dwords keeps 16B alignment and conflict-free banks.
// ---------------------------------------------------------------------------
#define APAD 20
#define BPAD 20

__launch_bounds__(256)
__global__ void gemm_bf16_kernel(const float* __restrict__ A,
                                 const float* __restrict__ W,
                                 const float* __restrict__ bias,
                                 float* __restrict__ C,
                                 int M, int N, int K, int act)
{
    __shared__ unsigned int As[128 * APAD];   // 128 rows x 16 pairs (+pad)
    __shared__ unsigned int Bs[128 * BPAD];   // 128 cols x 16 pairs (+pad)

    const int tid  = threadIdx.x;
    const int lane = tid & 31;
    const int wave = tid >> 5;
    const int wm   = wave & 3;        // 4 waves along M (32 rows each)
    const int wn   = wave >> 2;       // 2 waves along N (64 cols each)
    const int half = lane >> 4;
    const int l16  = lane & 15;

    const long blockM = (long)blockIdx.y * 128;
    const long blockN = (long)blockIdx.x * 128;

    floatx8 acc[2][4] = {};

    // A loader: thread -> (row, 16 consecutive k)
    const int ar = tid >> 1;            // 0..127
    const int ak = (tid & 1) * 16;      // element offset in k: 0 or 16
    // B loader: thread -> (k-pair, 8 consecutive cols)
    const int bkp = tid >> 4;           // 0..15  (k rows 2*bkp, 2*bkp+1)
    const int bnc = (tid & 15) * 8;     // 0..120

    for (int k0 = 0; k0 < K; k0 += 32) {
        // ---- stage A tile (128 x 32), branchless OOB via mask ----
        {
            const long gr  = blockM + ar;
            const int  vr  = (gr < M);
            const float* srow = A + (vr ? gr : 0) * (long)K + k0 + ak;
            const float4v* s4 = (const float4v*)srow;
            float4v f0 = s4[0], f1 = s4[1], f2 = s4[2], f3 = s4[3];
            const float msk = vr ? 1.0f : 0.0f;
            unsigned int* dst = &As[ar * APAD + (ak >> 1)];
            uint4v p0 = { pk2(f0.x*msk, f0.y*msk), pk2(f0.z*msk, f0.w*msk),
                          pk2(f1.x*msk, f1.y*msk), pk2(f1.z*msk, f1.w*msk) };
            uint4v p1 = { pk2(f2.x*msk, f2.y*msk), pk2(f2.z*msk, f2.w*msk),
                          pk2(f3.x*msk, f3.y*msk), pk2(f3.z*msk, f3.w*msk) };
            *(uint4v*)(dst)     = p0;
            *(uint4v*)(dst + 4) = p1;
            if (k0 + 32 < K && vr)
                __builtin_prefetch(srow + 32, 0, 1);    // global_prefetch_b8
        }
        // ---- stage B tile (32 x 128) transposed+packed ----
        {
            const float* r0 = W + (long)(k0 + 2 * bkp) * N + blockN + bnc;
            const float* r1 = r0 + N;
            const float4v* a4 = (const float4v*)r0;
            const float4v* b4 = (const float4v*)r1;
            float4v x0 = a4[0], x1 = a4[1];
            float4v y0 = b4[0], y1 = b4[1];
            unsigned int u[8] = {
                pk2(x0.x, y0.x), pk2(x0.y, y0.y), pk2(x0.z, y0.z), pk2(x0.w, y0.w),
                pk2(x1.x, y1.x), pk2(x1.y, y1.y), pk2(x1.z, y1.z), pk2(x1.w, y1.w) };
            #pragma unroll
            for (int c = 0; c < 8; ++c)
                Bs[(bnc + c) * BPAD + bkp] = u[c];
            if (k0 + 32 < K)
                __builtin_prefetch(r0 + 32L * N, 0, 1);
        }
        __syncthreads();

        // ---- fragments: 2x ds_load_b128 each ----
        union Frag { uint4v v[2]; bf16x16 f; };
        Frag fa[2], fb[4];
        #pragma unroll
        for (int i = 0; i < 2; ++i) {
            const unsigned int* p = &As[(wm * 32 + i * 16 + l16) * APAD + half * 4];
            fa[i].v[0] = *(const uint4v*)p;
            fa[i].v[1] = *(const uint4v*)(p + 8);
        }
        #pragma unroll
        for (int j = 0; j < 4; ++j) {
            const unsigned int* p = &Bs[(wn * 64 + j * 16 + l16) * BPAD + half * 4];
            fb[j].v[0] = *(const uint4v*)p;
            fb[j].v[1] = *(const uint4v*)(p + 8);
        }

        #pragma unroll
        for (int i = 0; i < 2; ++i)
            #pragma unroll
            for (int j = 0; j < 4; ++j)
                acc[i][j] = __builtin_amdgcn_wmma_f32_16x16x32_bf16(
                    false, fa[i].f, false, fb[j].f,
                    (short)0, acc[i][j], false, false);

        __syncthreads();
    }

    // ---- store (C/D layout: vgpr r -> M = half*8 + r, N = l16) ----
    #pragma unroll
    for (int i = 0; i < 2; ++i) {
        const long gm0 = blockM + wm * 32 + i * 16 + half * 8;
        #pragma unroll
        for (int j = 0; j < 4; ++j) {
            const long gn = blockN + wn * 64 + j * 16 + l16;
            float badd = bias ? bias[gn] : 0.0f;
            #pragma unroll
            for (int r = 0; r < 8; ++r) {
                const long gm = gm0 + r;
                if (gm < M && gn < N) {
                    float v = acc[i][j][r] + badd;
                    if (act) v = fmaxf(v, 0.0f);
                    C[gm * (long)N + gn] = v;
                }
            }
        }
    }
}

// ---------------------------------------------------------------------------
// Generic segmented attention. One block per (segment, head).
// qh/kh/vh rows are DMODEL wide (head-major h*64+d). Strides in rows.
// Lk of the last segment-in-batch may differ (ragged last block).
// Optionally writes probs (b, h, q, k) for the slf-last call.
// ---------------------------------------------------------------------------
__launch_bounds__(128)
__global__ void attn_kernel(const float* __restrict__ qh,
                            const float* __restrict__ kh,
                            const float* __restrict__ vh,
                            float* __restrict__ ctx,
                            float* __restrict__ probs,
                            int nq, int lk, int lk_last, int spb,
                            long q_bs, long q_ss,
                            long kv_bs, long kv_ss,
                            long c_bs, long c_ss, float scale)
{
    __shared__ float sc[512];
    __shared__ float qv[64];
    __shared__ float red[128];

    const int seg = blockIdx.x;
    const int h   = blockIdx.y;
    const int b   = seg / spb;
    const int si  = seg % spb;
    const int Lk  = (si == spb - 1) ? lk_last : lk;

    const float* qb = qh + (b * q_bs + si * q_ss)   * (long)DMODEL + h * DK;
    const float* kb = kh + (b * kv_bs + si * kv_ss) * (long)DMODEL + h * DK;
    const float* vb = vh + (b * kv_bs + si * kv_ss) * (long)DMODEL + h * DK;
    float*       cb = ctx + (b * c_bs + si * c_ss)  * (long)DMODEL + h * DK;

    const int tid = threadIdx.x;

    for (int q = 0; q < nq; ++q) {
        if (tid < 64) qv[tid] = qb[(long)q * DMODEL + tid];
        __syncthreads();

        for (int j = tid; j < Lk; j += 128) {
            const float* kr = kb + (long)j * DMODEL;
            float d = 0.0f;
            #pragma unroll 8
            for (int t = 0; t < 64; ++t) d += qv[t] * kr[t];
            sc[j] = d * scale;
        }
        __syncthreads();

        float m = -3.4e38f;
        for (int j = tid; j < Lk; j += 128) m = fmaxf(m, sc[j]);
        red[tid] = m; __syncthreads();
        for (int s = 64; s > 0; s >>= 1) {
            if (tid < s) red[tid] = fmaxf(red[tid], red[tid + s]);
            __syncthreads();
        }
        m = red[0]; __syncthreads();

        float ssum = 0.0f;
        for (int j = tid; j < Lk; j += 128) {
            float e = __expf(sc[j] - m);
            sc[j] = e; ssum += e;
        }
        red[tid] = ssum; __syncthreads();
        for (int s = 64; s > 0; s >>= 1) {
            if (tid < s) red[tid] += red[tid + s];
            __syncthreads();
        }
        const float inv = 1.0f / red[0];
        __syncthreads();
        for (int j = tid; j < Lk; j += 128) sc[j] *= inv;
        __syncthreads();

        if (probs) {
            for (int j = tid; j < Lk; j += 128)
                probs[(((long)b * NHEAD + h) * nq + q) * lk_last + j] = sc[j];
        }
        if (tid < 64) {
            float o = 0.0f;
            for (int j = 0; j < Lk; ++j)
                o += sc[j] * vb[(long)j * DMODEL + tid];
            cb[(long)q * DMODEL + tid] = o;
        }
        __syncthreads();
    }
}

// ---------------------------------------------------------------------------
// out = LayerNorm(x [+ bias] [+ res]) * g + beta   (row width DMODEL)
// ---------------------------------------------------------------------------
__launch_bounds__(256)
__global__ void ln_kernel(const float* __restrict__ x,
                          const float* __restrict__ res,
                          const float* __restrict__ bias,
                          const float* __restrict__ g,
                          const float* __restrict__ bet,
                          float* __restrict__ out)
{
    __shared__ float buf[DMODEL];
    __shared__ float red[256];
    const long r = blockIdx.x;
    const int tid = threadIdx.x;
    const float* xr = x + r * DMODEL;
    const float* rr = res ? res + r * DMODEL : nullptr;

    float s = 0.0f;
    for (int c = tid; c < DMODEL; c += 256) {
        float v = xr[c];
        if (bias) v += bias[c];
        if (rr)   v += rr[c];
        buf[c] = v; s += v;
    }
    red[tid] = s; __syncthreads();
    for (int st = 128; st > 0; st >>= 1) {
        if (tid < st) red[tid] += red[tid + st];
        __syncthreads();
    }
    const float mean = red[0] / (float)DMODEL;
    __syncthreads();
    float vs = 0.0f;
    for (int c = tid; c < DMODEL; c += 256) {
        float d = buf[c] - mean; vs += d * d;
    }
    red[tid] = vs; __syncthreads();
    for (int st = 128; st > 0; st >>= 1) {
        if (tid < st) red[tid] += red[tid + st];
        __syncthreads();
    }
    const float inv = rsqrtf(red[0] / (float)DMODEL + 1e-6f);
    for (int c = tid; c < DMODEL; c += 256)
        out[r * DMODEL + c] = (buf[c] - mean) * inv * g[c] + bet[c];
}

// dst[i] = src[(row % src_rows)][col] for DMODEL-wide rows
__global__ void bcast_kernel(const float* __restrict__ src,
                             float* __restrict__ dst,
                             int src_rows, long total)
{
    long i = (long)blockIdx.x * 256 + threadIdx.x;
    if (i >= total) return;
    long row = i >> 10;
    int  col = (int)(i & 1023);
    dst[i] = src[(row % src_rows) * (long)DMODEL + col];
}

// ---------------------------------------------------------------------------
// Host side
// ---------------------------------------------------------------------------
struct MhaP { const float *wq, *wk, *wv, *wo, *g, *b; };

static inline MhaP mk_mha(void* const* d_in, int base) {
    MhaP p;
    p.wq = (const float*)d_in[base + 0];
    p.wk = (const float*)d_in[base + 1];
    p.wv = (const float*)d_in[base + 2];
    p.wo = (const float*)d_in[base + 3];
    p.g  = (const float*)d_in[base + 4];
    p.b  = (const float*)d_in[base + 5];
    return p;
}

static inline void gemm(const float* A, const float* W, const float* bias,
                        float* C, int M, int N, int K, int act, hipStream_t s)
{
    dim3 grid((N + 127) / 128, (M + 127) / 128);
    gemm_bf16_kernel<<<grid, 256, 0, s>>>(A, W, bias, C, M, N, K, act);
}

static inline void bcast(const float* src, float* dst, int src_rows, long rows,
                         hipStream_t s)
{
    long total = rows * DMODEL;
    bcast_kernel<<<dim3((unsigned)((total + 255) / 256)), 256, 0, s>>>(
        src, dst, src_rows, total);
}

extern "C" void kernel_launch(void* const* d_in, const int* in_sizes, int n_in,
                              void* d_out, int out_size, void* d_ws, size_t ws_size,
                              hipStream_t stream)
{
    (void)in_sizes; (void)n_in; (void)out_size; (void)ws_size;
    const float* enc   = (const float*)d_in[0];   // (8,4100,1024)
    const float* aux_q = (const float*)d_in[1];   // (9,4,1024) = 36 rows
    const float* gws_q = (const float*)d_in[2];   // (1,4,1024) = 4 rows
    MhaP ax   = mk_mha(d_in, 3);
    MhaP gws  = mk_mha(d_in, 9);
    MhaP gwss = mk_mha(d_in, 15);
    MhaP axg  = mk_mha(d_in, 21);
    MhaP slf  = mk_mha(d_in, 27);
    const float* f_w1 = (const float*)d_in[33];
    const float* f_b1 = (const float*)d_in[34];
    const float* f_w2 = (const float*)d_in[35];
    const float* f_b2 = (const float*)d_in[36];
    const float* f_g  = (const float*)d_in[37];
    const float* f_b  = (const float*)d_in[38];

    float* out = (float*)d_out;                   // [NTF] enc_output + [2048] attn_last
    float* probs_out = out + NTF;

    float* ws = (float*)d_ws;
    float* A_buf = ws;                            // NT x 1024  (kh_ax -> qh_slf)
    float* B_buf = A_buf + NTF;                   // NT x 1024  (vh_ax -> ctx_slf)
    float* C_buf = B_buf + NTF;                   // NT x 1024  (enc_mid)
    float* H_buf = C_buf + NTF;                   // 4096 x 4096 (FFN chunk)
    float* S = H_buf + 4096L * 4096L;
    long o = 0;
    auto alloc = [&](long rows) { float* p = S + o * DMODEL; o += rows; return p; };
    float* aux_qh   = alloc(36);
    float* qh_rep   = alloc(288);
    float* ctx_ax   = alloc(288);
    float* tmp288   = alloc(288);
    float* all_sl   = alloc(288);
    float* qres_aux = alloc(288);
    float* qg       = alloc(4);
    float* qg_rep   = alloc(32);
    float* khg      = alloc(288);
    float* vhg      = alloc(288);
    float* ctxg     = alloc(32);
    float* tmpg     = alloc(32);
    float* gqres    = alloc(32);
    float* g1       = alloc(32);
    float* q2       = alloc(32);
    float* k2       = alloc(32);
    float* v2       = alloc(32);
    float* ctx2     = alloc(32);
    float* tmp2     = alloc(32);
    float* g2       = alloc(32);
    float* q3       = alloc(288);
    float* k3       = alloc(32);
    float* v3       = alloc(32);
    float* ctx3     = alloc(288);
    float* tmp3     = alloc(288);
    float* slots2   = alloc(288);
    float* khs      = alloc(288);
    float* vhs      = alloc(288);

    const float scale = 0.125f;   // 1/sqrt(64)

    // ---------------- Stage A: ax cross-attention -> all_slots ----------------
    gemm(enc, ax.wk, nullptr, A_buf, NT, DMODEL, DMODEL, 0, stream);
    gemm(enc, ax.wv, nullptr, B_buf, NT, DMODEL, DMODEL, 0, stream);
    gemm(aux_q, ax.wq, nullptr, aux_qh, 36, DMODEL, DMODEL, 0, stream);
    bcast(aux_qh, qh_rep, 36, 288, stream);
    attn_kernel<<<dim3(72, NHEAD), 128, 0, stream>>>(
        qh_rep, A_buf, B_buf, ctx_ax, nullptr,
        NSLOT, BS, 4, NBLK,
        36, 4, TLEN, BS, 36, 4, scale);
    gemm(ctx_ax, ax.wo, nullptr, tmp288, 288, DMODEL, DMODEL, 0, stream);
    bcast(aux_q, qres_aux, 36, 288, stream);
    ln_kernel<<<288, 256, 0, stream>>>(tmp288, qres_aux, nullptr, ax.g, ax.b, all_sl);

    // ---------------- Stage B1: gws ----------------
    gemm(gws_q, gws.wq, nullptr, qg, 4, DMODEL, DMODEL, 0, stream);
    bcast(qg, qg_rep, 4, 32, stream);
    gemm(all_sl, gws.wk, nullptr, khg, 288, DMODEL, DMODEL, 0, stream);
    gemm(all_sl, gws.wv, nullptr, vhg, 288, DMODEL, DMODEL, 0, stream);
    attn_kernel<<<dim3(8, NHEAD), 128, 0, stream>>>(
        qg_rep, khg, vhg, ctxg, nullptr,
        NSLOT, 36, 36, 1, 4, 0, 36, 0, 4, 0, scale);
    gemm(ctxg, gws.wo, nullptr, tmpg, 32, DMODEL, DMODEL, 0, stream);
    bcast(gws_q, gqres, 4, 32, stream);
    ln_kernel<<<32, 256, 0, stream>>>(tmpg, gqres, nullptr, gws.g, gws.b, g1);

    // ---------------- Stage B2: gws_self ----------------
    gemm(g1, gwss.wq, nullptr, q2, 32, DMODEL, DMODEL, 0, stream);
    gemm(g1, gwss.wk, nullptr, k2, 32, DMODEL, DMODEL, 0, stream);
    gemm(g1, gwss.wv, nullptr, v2, 32, DMODEL, DMODEL, 0, stream);
    attn_kernel<<<dim3(8, NHEAD), 128, 0, stream>>>(
        q2, k2, v2, ctx2, nullptr,
        NSLOT, NSLOT, NSLOT, 1, 4, 0, 4, 0, 4, 0, scale);
    gemm(ctx2, gwss.wo, nullptr, tmp2, 32, DMODEL, DMODEL, 0, stream);
    ln_kernel<<<32, 256, 0, stream>>>(tmp2, g1, nullptr, gwss.g, gwss.b, g2);

    // ---------------- Stage B3: ax_gws (slots attend to g2) ----------------
    gemm(all_sl, axg.wq, nullptr, q3, 288, DMODEL, DMODEL, 0, stream);
    gemm(g2, axg.wk, nullptr, k3, 32, DMODEL, DMODEL, 0, stream);
    gemm(g2, axg.wv, nullptr, v3, 32, DMODEL, DMODEL, 0, stream);
    attn_kernel<<<dim3(8, NHEAD), 128, 0, stream>>>(
        q3, k3, v3, ctx3, nullptr,
        36, NSLOT, NSLOT, 1, 36, 0, 4, 0, 36, 0, scale);
    gemm(ctx3, axg.wo, nullptr, tmp3, 288, DMODEL, DMODEL, 0, stream);
    ln_kernel<<<288, 256, 0, stream>>>(tmp3, all_sl, nullptr, axg.g, axg.b, slots2);

    // ---------------- Stage C: slf (tokens attend to their slots) ----------------
    gemm(enc, slf.wq, nullptr, A_buf, NT, DMODEL, DMODEL, 0, stream);
    gemm(slots2, slf.wk, nullptr, khs, 288, DMODEL, DMODEL, 0, stream);
    gemm(slots2, slf.wv, nullptr, vhs, 288, DMODEL, DMODEL, 0, stream);
    attn_kernel<<<dim3(64, NHEAD), 128, 0, stream>>>(
        A_buf, khs, vhs, B_buf, nullptr,
        BS, NSLOT, NSLOT, NM,
        TLEN, BS, 36, 4, TLEN, BS, scale);
    attn_kernel<<<dim3(8, NHEAD), 128, 0, stream>>>(
        A_buf + 4096L * DMODEL, khs + 32L * DMODEL, vhs + 32L * DMODEL,
        B_buf + 4096L * DMODEL, probs_out,
        NSLOT, NSLOT, NSLOT, 1,
        TLEN, 0, 36, 0, TLEN, 0, scale);
    gemm(B_buf, slf.wo, nullptr, out, NT, DMODEL, DMODEL, 0, stream);
    ln_kernel<<<NT, 256, 0, stream>>>(out, enc, nullptr, slf.g, slf.b, C_buf);

    // ---------------- Stage D: FFN, chunked over M ----------------
    const long CH = 4096;
    for (long m0 = 0; m0 < NT; m0 += CH) {
        int mr = (int)((NT - m0) < CH ? (NT - m0) : CH);
        gemm(C_buf + m0 * DMODEL, f_w1, f_b1, H_buf, mr, DINNER, DMODEL, 1, stream);
        gemm(H_buf, f_w2, nullptr, out + m0 * DMODEL, mr, DMODEL, DINNER, 0, stream);
    }
    ln_kernel<<<NT, 256, 0, stream>>>(out, C_buf, f_b2, f_g, f_b, out);
}
